// LSTM_14405320311388
// MI455X (gfx1250) — compile-verified, matrix operand order checked by reading, not run on hardware
//
#include <hip/hip_runtime.h>
#include <hip/hip_bf16.h>
#include <math.h>

// Problem constants
#define B_    128
#define T_    512
#define I_    300
#define H_    256
#define G_    1024            // 4*H
#define NT_   64              // G/16 gate-column tiles
#define KC_IH 10              // ceil(300/32) k-chunks (zero padded to 320)
#define KC_HH 8               // 256/32 k-chunks

typedef __attribute__((ext_vector_type(16))) __bf16 v16bf;
typedef __attribute__((ext_vector_type(8)))  __bf16 bf16x8;
typedef __attribute__((ext_vector_type(4)))  __bf16 bf16x4;
typedef __attribute__((ext_vector_type(8)))  float  v8f;

union ABop { v16bf v; bf16x8 h[2]; };

// Scheduler pinning: 0x8 = matrix ops, 0x20 = VMEM read, 0x100 = DS read
#if __has_builtin(__builtin_amdgcn_sched_group_barrier)
#define SCHED_GROUP(m, s, i) __builtin_amdgcn_sched_group_barrier((m), (s), (i))
#else
#define SCHED_GROUP(m, s, i)
#endif

// K-offset inside a 16x32 bf16 WMMA tile for (lane, element) per CDNA5 ISA:
// lanes 0-15 hold K {0..7,16..23}, lanes 16-31 hold K {8..15,24..31}
__device__ __forceinline__ int koff16(int lane, int e) {
  return (e & 7) + ((e >> 3) << 4) + ((lane >> 4) << 3);
}
__device__ __forceinline__ float rcpf_(float x) {
#if __has_builtin(__builtin_amdgcn_rcpf)
  return __builtin_amdgcn_rcpf(x);      // v_rcp_f32, single instruction
#else
  return 1.0f / x;
#endif
}
__device__ __forceinline__ float sigmoidf_(float x) {
  return rcpf_(1.0f + __expf(-x));      // v_exp + v_add + v_rcp
}
__device__ __forceinline__ float tanhf_(float x) {
  // tanh(x) = 1 - 2/(exp(2x)+1); saturates correctly at +/-inf
  return 1.0f - 2.0f * rcpf_(__expf(2.0f * x) + 1.0f);
}
__device__ __forceinline__ v8f vzero8() {
  v8f v;
#pragma unroll
  for (int e = 0; e < 8; ++e) v[e] = 0.0f;
  return v;
}

// ---------------------------------------------------------------------------
// Pack a (1024 x K) fp32 row-major weight into bf16 WMMA-B tile-major layout:
// dst[((tile*KC + kc)*32 + lane)*16 + e] = W[tile*16 + lane%16, kc*32 + koff]
// so the GEMM kernels load one contiguous 32B v16bf per lane per tile.
// ---------------------------------------------------------------------------
__global__ __launch_bounds__(256) void pack_kernel(const float* __restrict__ Wsrc,
                                                   __bf16* __restrict__ dst,
                                                   int K, int KC) {
  int idx = blockIdx.x * blockDim.x + threadIdx.x;
  int total = NT_ * KC * 512;
  if (idx >= total) return;
  int e    = idx & 15;
  int lane = (idx >> 4) & 31;
  int kc   = (idx >> 9) % KC;
  int tile = (idx >> 9) / KC;
  int n = tile * 16 + (lane & 15);
  int k = kc * 32 + koff16(lane, e);
  int kk = (k < K) ? k : (K - 1);
  float f = Wsrc[(size_t)n * K + kk];
  dst[idx] = (__bf16)((k < K) ? f : 0.0f);
}

// ---------------------------------------------------------------------------
// Phase 1: forward gate pre-activations
//   xp[m, j, g] = sum_k x[m, k] * W_ih_f[g*256+j, k] + b_f[g*256+j]
// bf16, GATE-INTERLEAVED so the scan reads one bf16x4 per (row, hidden unit).
// Workgroup tile: 32 (M) x 512 (N); 8 waves; K staged through LDS as bf16.
// ---------------------------------------------------------------------------
__global__ __launch_bounds__(256) void proj_kernel(const float* __restrict__ x,
                                                   const __bf16* __restrict__ wpk,
                                                   const float* __restrict__ bias,
                                                   __bf16* __restrict__ xp) {
  __shared__ __bf16 xs[32 * 32];
  const int tid   = threadIdx.x;
  const int lane  = tid & 31;
  const int w     = tid >> 5;                      // wave 0..7
  const int mblk  = blockIdx.y * 32;               // first of 32 rows (b*T+t)
  const int nt0   = blockIdx.x * 32 + (w & 3) * 8; // first of 8 N-tiles
  const int mhalf = w >> 2;                        // which 16-row tile
  const int mhi   = (lane >> 4) << 3;
  const int nl    = lane & 15;

  v8f acc[8];
#pragma unroll
  for (int j = 0; j < 8; ++j) acc[j] = vzero8();

  const int srow = tid >> 3;
  const int skq  = (tid & 7) * 4;
  const float* xrow = x + (size_t)(mblk + srow) * I_;

#pragma unroll 1
  for (int kc = 0; kc < KC_IH; ++kc) {
    __syncthreads();
    {   // cooperative stage: 32 rows x 32 k, fp32 -> bf16
      const int k0 = kc * 32 + skq;
      bf16x4 vv;
      if (kc < 9) {                     // fully in-bounds: one 16B load
        const float4 f4 = *reinterpret_cast<const float4*>(xrow + k0);
        vv[0] = (__bf16)f4.x; vv[1] = (__bf16)f4.y;
        vv[2] = (__bf16)f4.z; vv[3] = (__bf16)f4.w;
      } else {                          // tail chunk: clamp + select, no branch
#pragma unroll
        for (int j = 0; j < 4; ++j) {
          int k  = k0 + j;
          int kk = (k < I_) ? k : (I_ - 1);
          float f = xrow[kk];
          vv[j] = (__bf16)((k < I_) ? f : 0.0f);
        }
      }
      *reinterpret_cast<bf16x4*>(&xs[srow * 32 + skq]) = vv;
    }
    __syncthreads();
    ABop a;
    {   // A operand in WMMA layout straight from LDS (2x ds_load_b128)
      const int row = mhalf * 16 + nl;
      a.h[0] = *reinterpret_cast<const bf16x8*>(&xs[row * 32 + mhi]);
      a.h[1] = *reinterpret_cast<const bf16x8*>(&xs[row * 32 + mhi + 16]);
    }
    v16bf bops[8];
#pragma unroll
    for (int j = 0; j < 8; ++j)
      bops[j] = *reinterpret_cast<const v16bf*>(
          wpk + ((size_t)(nt0 + j) * KC_IH + kc) * 512 + lane * 16);
#pragma unroll
    for (int j = 0; j < 8; ++j)
      acc[j] = __builtin_amdgcn_wmma_f32_16x16x32_bf16(false, a.v, false, bops[j],
                                                       (short)0, acc[j], false, false);
    // pin schedule: A ds-reads, then ALL 16 B vmem-reads, then the 8 WMMAs
    SCHED_GROUP(0x100, 2, 0);
    SCHED_GROUP(0x20, 16, 0);
    SCHED_GROUP(0x8, 8, 0);
  }
  // store xp (bias folded) bf16, gate-interleaved [m][j][g]
#pragma unroll
  for (int j = 0; j < 8; ++j) {
    const int ncol = (nt0 + j) * 16 + nl;       // gate column 0..1023
    const int gidx = ncol >> 8;                 // which gate (i,f,g,o)
    const int jcol = ncol & 255;                // hidden unit
    const float bv = bias[ncol];
#pragma unroll
    for (int r = 0; r < 8; ++r) {
      const int m = mblk + mhalf * 16 + r + mhi;
      xp[((size_t)m * 256 + jcol) * 4 + gidx] = (__bf16)(acc[j][r] + bv);
    }
  }
}

// ---------------------------------------------------------------------------
// Phase 2: forward LSTM scan. 8 persistent WGs x 512 threads (16 waves);
// each WG owns 16 batch rows, wave w owns hidden columns [16w, 16w+16)
// -> gate tiles {g*16 + w}: all four gates of a hidden unit live in the
// same wave/slot. Cell state c stays in VGPRs for all 512 steps; h ping-
// pongs through LDS as bf16 in A-operand-ready form (2x ds_load_b128).
// The k-loop is manually 2x unrolled and sched-group pinned so chunk kc+1's
// 8 global_load_b128 are in flight while chunk kc's WMMAs execute.
// ---------------------------------------------------------------------------
__global__ __launch_bounds__(512) void scan_kernel(const __bf16* __restrict__ xp,
                                                   const __bf16* __restrict__ whhpk,
                                                   float* __restrict__ hlast) {
  __shared__ __bf16 hbuf[2][16 * 256];
  const int tid  = threadIdx.x;
  const int lane = tid & 31;
  const int w    = tid >> 5;            // wave 0..15, owns j-tile w
  const int bt0  = blockIdx.x * 16;
  const int mhi  = (lane >> 4) << 3;
  const int nl   = lane & 15;
  const int j    = w * 16 + nl;         // this lane's hidden column

  for (int i = tid; i < 16 * 256; i += 512) hbuf[0][i] = (__bf16)0.0f;
  __syncthreads();

  v8f c  = vzero8();
  v8f hv = vzero8();

#pragma unroll 1
  for (int t = 0; t < T_; ++t) {
    const __bf16* hr = hbuf[t & 1];
    __bf16* hw = hbuf[(t & 1) ^ 1];

    // issue this step's xp loads early; latency hides under the WMMAs,
    // and warm next step's lines with global_prefetch (no counter cost)
    bf16x4 xv[8];
#pragma unroll
    for (int r = 0; r < 8; ++r) {
      const size_t m = (size_t)(bt0 + r + mhi) * T_ + t;
      xv[r] = *reinterpret_cast<const bf16x4*>(&xp[(m * 256 + j) * 4]);
      __builtin_prefetch(&xp[((m + 1) * 256 + j) * 4], 0, 3);
    }

    v8f acc[4];
#pragma unroll
    for (int g = 0; g < 4; ++g) acc[g] = vzero8();

#pragma unroll 1
    for (int kc = 0; kc < KC_HH; kc += 2) {
      ABop a0, a1;
      v16bf b0[4], b1[4];
      {
        const int off0 = kc * 32 + mhi;
        a0.h[0] = *reinterpret_cast<const bf16x8*>(&hr[nl * 256 + off0]);
        a0.h[1] = *reinterpret_cast<const bf16x8*>(&hr[nl * 256 + off0 + 16]);
        a1.h[0] = *reinterpret_cast<const bf16x8*>(&hr[nl * 256 + off0 + 32]);
        a1.h[1] = *reinterpret_cast<const bf16x8*>(&hr[nl * 256 + off0 + 48]);
      }
#pragma unroll
      for (int g = 0; g < 4; ++g) {
        const size_t tb = ((size_t)(g * 16 + w) * KC_HH + kc) * 512 + lane * 16;
        b0[g] = *reinterpret_cast<const v16bf*>(whhpk + tb);
        b1[g] = *reinterpret_cast<const v16bf*>(whhpk + tb + 512);
      }
#pragma unroll
      for (int g = 0; g < 4; ++g)
        acc[g] = __builtin_amdgcn_wmma_f32_16x16x32_bf16(
            false, a0.v, false, b0[g], (short)0, acc[g], false, false);
#pragma unroll
      for (int g = 0; g < 4; ++g)
        acc[g] = __builtin_amdgcn_wmma_f32_16x16x32_bf16(
            false, a1.v, false, b1[g], (short)0, acc[g], false, false);
      // all loads (both chunks) ahead of all 8 WMMAs
      SCHED_GROUP(0x100, 4, 0);
      SCHED_GROUP(0x20, 16, 0);
      SCHED_GROUP(0x8, 8, 0);
    }
#pragma unroll
    for (int r = 0; r < 8; ++r) {
      float gi = acc[0][r] + (float)xv[r][0];
      float gf = acc[1][r] + (float)xv[r][1];
      float gg = acc[2][r] + (float)xv[r][2];
      float go = acc[3][r] + (float)xv[r][3];
      float cn = sigmoidf_(gf) * c[r] + sigmoidf_(gi) * tanhf_(gg);
      c[r] = cn;
      float hn = sigmoidf_(go) * tanhf_(cn);
      hv[r] = hn;
      hw[(r + mhi) * 256 + j] = (__bf16)hn;
    }
    __syncthreads();
  }
#pragma unroll
  for (int r = 0; r < 8; ++r)
    hlast[(size_t)(bt0 + r + mhi) * H_ + j] = hv[r];
}

// ---------------------------------------------------------------------------
// Phase 3: backward direction collapses to ONE step from h0=c0=0 (so the
// W_hh_b term is exactly zero): g = x[:,T-1] @ W_ih_b^T + b_b, then gates,
// then the 2-wide output head fused in. 16 waves, wave w owns j-tile w.
// ---------------------------------------------------------------------------
__global__ __launch_bounds__(512) void bwd_head_kernel(const float* __restrict__ x,
                                                       const __bf16* __restrict__ wpk,
                                                       const float* __restrict__ bb,
                                                       const float* __restrict__ hf,
                                                       const float* __restrict__ wlin,
                                                       const float* __restrict__ blin,
                                                       float* __restrict__ out) {
  __shared__ float hbl[16 * 256];
  const int tid  = threadIdx.x;
  const int lane = tid & 31;
  const int w    = tid >> 5;            // 0..15
  const int bt0  = blockIdx.x * 16;
  const int mhi  = (lane >> 4) << 3;
  const int nl   = lane & 15;
  const int j    = w * 16 + nl;

  v8f acc[4];
#pragma unroll
  for (int g = 0; g < 4; ++g) acc[g] = vzero8();

  const float* xrow = x + ((size_t)(bt0 + nl) * T_ + (T_ - 1)) * I_;
#pragma unroll 1
  for (int kc = 0; kc < KC_IH; ++kc) {
    ABop a;
#pragma unroll
    for (int e = 0; e < 16; ++e) {
      int k  = kc * 32 + koff16(lane, e);
      int kk = (k < I_) ? k : (I_ - 1);
      float f = xrow[kk];
      a.v[e] = (__bf16)((k < I_) ? f : 0.0f);
    }
    v16bf bops[4];
#pragma unroll
    for (int g = 0; g < 4; ++g)
      bops[g] = *reinterpret_cast<const v16bf*>(
          wpk + ((size_t)(g * 16 + w) * KC_IH + kc) * 512 + lane * 16);
#pragma unroll
    for (int g = 0; g < 4; ++g)
      acc[g] = __builtin_amdgcn_wmma_f32_16x16x32_bf16(
          false, a.v, false, bops[g], (short)0, acc[g], false, false);
  }
#pragma unroll
  for (int r = 0; r < 8; ++r) {
    float gi = acc[0][r] + bb[j];
    float gg = acc[2][r] + bb[2 * H_ + j];
    float go = acc[3][r] + bb[3 * H_ + j];
    float cn = sigmoidf_(gi) * tanhf_(gg);          // f-gate * c0 == 0
    hbl[(r + mhi) * 256 + j] = sigmoidf_(go) * tanhf_(cn);
  }
  __syncthreads();
  if (tid < 32) {                                    // 16 rows x 2 outputs
    const int b = nl;
    const int o = lane >> 4;
    float s = blin[o];
    const float* hfr = hf + (size_t)(bt0 + b) * H_;
#pragma unroll 1
    for (int jj = 0; jj < H_; ++jj) {
      s += hfr[jj] * wlin[o * 512 + jj];
      s += hbl[b * 256 + jj] * wlin[o * 512 + H_ + jj];
    }
    out[(bt0 + b) * 2 + o] = s;
  }
}

// ---------------------------------------------------------------------------
extern "C" void kernel_launch(void* const* d_in, const int* in_sizes, int n_in,
                              void* d_out, int out_size, void* d_ws, size_t ws_size,
                              hipStream_t stream) {
  (void)in_sizes; (void)n_in; (void)out_size; (void)ws_size;
  const float* x    = (const float*)d_in[0];
  const float* Wihf = (const float*)d_in[1];
  const float* Whhf = (const float*)d_in[2];
  const float* bfv  = (const float*)d_in[3];
  const float* Wihb = (const float*)d_in[4];
  /* d_in[5] = W_hh_b: provably unused (backward state starts at zero) */
  const float* bbv  = (const float*)d_in[6];
  const float* Wlin = (const float*)d_in[7];
  const float* blin = (const float*)d_in[8];
  float* out = (float*)d_out;

  char* ws = (char*)d_ws;
  __bf16* xp      = (__bf16*)(ws);                                   // 134,217,728 B
  __bf16* wihf_pk = (__bf16*)(ws + 134217728);                       //     655,360 B
  __bf16* wihb_pk = (__bf16*)(ws + 134217728 + 655360);              //     655,360 B
  __bf16* whhf_pk = (__bf16*)(ws + 134217728 + 2 * 655360);          //     524,288 B
  float*  hfl     = (float*) (ws + 134217728 + 2 * 655360 + 524288); //     131,072 B

  pack_kernel<<<(NT_ * KC_IH * 512 + 255) / 256, 256, 0, stream>>>(Wihf, wihf_pk, I_, KC_IH);
  pack_kernel<<<(NT_ * KC_IH * 512 + 255) / 256, 256, 0, stream>>>(Wihb, wihb_pk, I_, KC_IH);
  pack_kernel<<<(NT_ * KC_HH * 512 + 255) / 256, 256, 0, stream>>>(Whhf, whhf_pk, H_, KC_HH);

  proj_kernel<<<dim3(2, (B_ * T_) / 32), 256, 0, stream>>>(x, wihf_pk, bfv, xp);
  scan_kernel<<<B_ / 16, 512, 0, stream>>>(xp, whhf_pk, hfl);
  bwd_head_kernel<<<B_ / 16, 512, 0, stream>>>(x, wihb_pk, bbv, hfl, Wlin, blin, out);
}
